// Model_47837345742971
// MI455X (gfx1250) — matrix operand order repaired and checked by reading, not compiled, not run on hardware
//
#include <hip/hip_runtime.h>

// Problem constants
#define Bb   4096
#define Tt   256
#define Dd   64
#define Hh   128
#define Aa   8
#define G4H  512      // 4*H
#define KK   192      // D + H (packed K dimension)
#define NCH  6        // K chunks of 32
#define ASTR 196      // LDS row stride in halves (conflict-avoiding pad)

typedef __attribute__((ext_vector_type(16))) _Float16 v16h;
typedef __attribute__((ext_vector_type(8)))  float    v8f;

union HFrag { v16h h; unsigned int u[8]; };

// ---------------------------------------------------------------------------
// Pre-pass: pack W_ih (4H x D) and W_hh (4H x H) into one fp16 row-major
// [n=512][k=192] buffer (k<64 -> W_ih, k>=64 -> W_hh), and fuse the biases.
// ---------------------------------------------------------------------------
__global__ __launch_bounds__(192)
void pack_weights_kernel(const float* __restrict__ Wih, const float* __restrict__ Whh,
                         const float* __restrict__ bih, const float* __restrict__ bhh,
                         _Float16* __restrict__ Wpk, float* __restrict__ bias) {
  const int n = blockIdx.x;     // 0..511
  const int k = threadIdx.x;    // 0..191
  float w = (k < Dd) ? Wih[n * Dd + k] : Whh[n * Hh + (k - Dd)];
  Wpk[n * KK + k] = (_Float16)w;
  if (k == 0) bias[n] = bih[n] + bhh[n];
}

__device__ __forceinline__ float fsig(float x) {
  return __builtin_amdgcn_rcpf(1.0f + __expf(-x));
}
__device__ __forceinline__ float ftanh(float x) {
  return 2.0f * __builtin_amdgcn_rcpf(1.0f + __expf(-2.0f * x)) - 1.0f;
}

// ---------------------------------------------------------------------------
// Main kernel: one block = 16 batch rows, 8 waves. Full T-step LSTM with
// register-resident weights (B fragments) and cell state, then MLP heads.
// ---------------------------------------------------------------------------
__global__ __launch_bounds__(256, 1)
void lstm_policy_kernel(const float* __restrict__ state,
                        const _Float16* __restrict__ Wpk,
                        const float* __restrict__ bias,
                        const float* __restrict__ mu_w1,  const float* __restrict__ mu_b1,
                        const float* __restrict__ mu_w2,  const float* __restrict__ mu_b2,
                        const float* __restrict__ var_w1, const float* __restrict__ var_b1,
                        const float* __restrict__ var_w2, const float* __restrict__ var_b2,
                        const float* __restrict__ val_w1, const float* __restrict__ val_b1,
                        const float* __restrict__ val_w2, const float* __restrict__ val_b2,
                        float* __restrict__ out_mu, float* __restrict__ out_var,
                        float* __restrict__ out_val) {
  __shared__ _Float16 Apad[16][ASTR];        // rows = batch tile, cols = K (x || h), fp16
  __shared__ float    hid[3][16][Hh / 2];    // head layer-1 activations

  const int tid    = threadIdx.x;
  const int wave   = tid >> 5;
  const int lane   = tid & 31;
  const int lrow   = lane & 15;
  const int lhalf  = lane >> 4;
  const int batch0 = blockIdx.x * 16;

  // --- Load this wave's B fragments into registers (4 gates x 6 K-chunks).
  // B 16-bit 32x16 layout: lane = column N (0-15 | 0-15), VGPR0-3: K = 8*lhalf + 2v,
  // VGPR4-7: K = 16 + 8*lhalf + 2(v-4).
  HFrag bfrag[4][NCH];
  #pragma unroll
  for (int g = 0; g < 4; ++g) {
    const int n = g * Hh + wave * 16 + lrow;   // output column (gate-major)
    const unsigned int* rowp = (const unsigned int*)(Wpk + (size_t)n * KK);
    #pragma unroll
    for (int c = 0; c < NCH; ++c) {
      const unsigned int* p = rowp + c * 16;   // 32 halves = 16 dwords per chunk
      #pragma unroll
      for (int v = 0; v < 8; ++v) bfrag[g][c].u[v] = p[(v < 4) ? (4 * lhalf + v) : (8 + 4 * lhalf + (v - 4))];
    }
  }

  float bz[4];
  #pragma unroll
  for (int g = 0; g < 4; ++g) bz[g] = bias[g * Hh + wave * 16 + lrow];

  // Zero the h region of the staging buffer (h_0 = 0).
  for (int i = tid; i < 16 * Hh; i += 256)
    Apad[i >> 7][Dd + (i & 127)] = (_Float16)0.0f;

  float cst[8];
  #pragma unroll
  for (int e = 0; e < 8; ++e) cst[e] = 0.0f;

  const float* srow = state + (size_t)batch0 * Tt * Dd;

  for (int t = 0; t < Tt; ++t) {
    // Stage x_t (16 x 64 fp32 -> fp16) into Apad[:, 0:64). Coalesced: 256B/row.
    #pragma unroll
    for (int i = tid; i < 16 * Dd; i += 256) {
      const int r = i >> 6, d = i & 63;
      Apad[r][d] = (_Float16)srow[(size_t)r * Tt * Dd + (size_t)t * Dd + d];
    }
    __syncthreads();   // x_t staged, h_{t-1} visible

    v8f acc[4];
    #pragma unroll
    for (int g = 0; g < 4; ++g) {
      #pragma unroll
      for (int e = 0; e < 8; ++e) acc[g][e] = bz[g];
    }

    // z = [x_t | h] @ [W_ih | W_hh]^T + b : 6 K-chunks, A reused across 4 gates.
    #pragma unroll
    for (int c = 0; c < NCH; ++c) {
      // A 16-bit 16x32 layout: lane = row M, VGPR0-3: K = 8*lhalf + 2v,
      // VGPR4-7: K = 16 + 8*lhalf + 2(v-4).
      HFrag a;
      const unsigned int* q = (const unsigned int*)(&Apad[lrow][c * 32]);
      #pragma unroll
      for (int v = 0; v < 4; ++v) a.u[v] = q[4 * lhalf + v];
      #pragma unroll
      for (int v = 4; v < 8; ++v) a.u[v] = q[8 + 4 * lhalf + (v - 4)];
      #pragma unroll
      for (int g = 0; g < 4; ++g)
        acc[g] = __builtin_amdgcn_wmma_f32_16x16x32_f16(
            false, a.h, false, bfrag[g][c].h, (short)0, acc[g], false, false);
    }

    // Gates + cell update, fully register-local (i,f,g,o share lane/VGPR slot).
    float hv[8];
    #pragma unroll
    for (int e = 0; e < 8; ++e) {
      const float ig = fsig(acc[0][e]);
      const float fg = fsig(acc[1][e]);
      const float gg = ftanh(acc[2][e]);
      const float og = fsig(acc[3][e]);
      const float cn = fg * cst[e] + ig * gg;
      cst[e] = cn;
      hv[e]  = og * ftanh(cn);
    }

    __syncthreads();   // everyone done reading Apad before h overwrite
    #pragma unroll
    for (int e = 0; e < 8; ++e)   // C/D layout: row M = e + 8*lhalf, col = lane%16
      Apad[e + 8 * lhalf][Dd + wave * 16 + lrow] = (_Float16)hv[e];
  }
  __syncthreads();   // final h visible to all threads

  // ---------------- MLP heads (tiny: ~0.2 GFLOP total) ----------------
  // Layer 1: 3 heads x 16 rows x 64 units, dot over H=128.
  for (int i = tid; i < 3 * 16 * 64; i += 256) {
    const int head = i >> 10;
    const int rem  = i & 1023;
    const int r    = rem >> 6;
    const int u    = rem & 63;
    const float* w1 = (head == 0) ? mu_w1 : (head == 1) ? var_w1 : val_w1;
    const float* b1 = (head == 0) ? mu_b1 : (head == 1) ? var_b1 : val_b1;
    const float* wr = w1 + u * Hh;
    float s = b1[u];
    #pragma unroll 8
    for (int j = 0; j < Hh; ++j) s += (float)Apad[r][Dd + j] * wr[j];
    hid[head][r][u] = fmaxf(s, 0.0f);
  }
  __syncthreads();

  // Layer 2: mu (8) + var (8) + value (1) per row.
  for (int i = tid; i < 16 * 17; i += 256) {
    const int r = i / 17;
    const int u = i % 17;
    if (u < Aa) {
      const float* wr = mu_w2 + u * (Hh / 2);
      float s = mu_b2[u];
      #pragma unroll 8
      for (int j = 0; j < Hh / 2; ++j) s += hid[0][r][j] * wr[j];
      out_mu[(size_t)(batch0 + r) * Aa + u] = tanhf(s);
    } else if (u < 2 * Aa) {
      const int uu = u - Aa;
      const float* wr = var_w2 + uu * (Hh / 2);
      float s = var_b2[uu];
      #pragma unroll 8
      for (int j = 0; j < Hh / 2; ++j) s += hid[1][r][j] * wr[j];
      // softplus, numerically stable
      out_var[(size_t)(batch0 + r) * Aa + uu] = fmaxf(s, 0.0f) + log1pf(__expf(-fabsf(s)));
    } else {
      float s = val_b2[0];
      #pragma unroll 8
      for (int j = 0; j < Hh / 2; ++j) s += hid[2][r][j] * val_w2[j];
      out_val[batch0 + r] = s;
    }
  }
}

// ---------------------------------------------------------------------------
extern "C" void kernel_launch(void* const* d_in, const int* in_sizes, int n_in,
                              void* d_out, int out_size, void* d_ws, size_t ws_size,
                              hipStream_t stream) {
  const float* state  = (const float*)d_in[0];
  const float* W_ih   = (const float*)d_in[1];
  const float* W_hh   = (const float*)d_in[2];
  const float* b_ih   = (const float*)d_in[3];
  const float* b_hh   = (const float*)d_in[4];
  const float* mu_w1  = (const float*)d_in[5];
  const float* mu_b1  = (const float*)d_in[6];
  const float* mu_w2  = (const float*)d_in[7];
  const float* mu_b2  = (const float*)d_in[8];
  const float* var_w1 = (const float*)d_in[9];
  const float* var_b1 = (const float*)d_in[10];
  const float* var_w2 = (const float*)d_in[11];
  const float* var_b2 = (const float*)d_in[12];
  const float* val_w1 = (const float*)d_in[13];
  const float* val_b1 = (const float*)d_in[14];
  const float* val_w2 = (const float*)d_in[15];
  const float* val_b2 = (const float*)d_in[16];

  _Float16* Wpk = (_Float16*)d_ws;
  float* bias = (float*)((char*)d_ws + (size_t)G4H * KK * sizeof(_Float16));

  pack_weights_kernel<<<G4H, KK, 0, stream>>>(W_ih, W_hh, b_ih, b_hh, Wpk, bias);

  float* out = (float*)d_out;
  float* out_mu  = out;                         // [B, 8]
  float* out_var = out + (size_t)Bb * Aa;       // [B, 8]
  float* out_val = out + (size_t)Bb * Aa * 2;   // [B, 1]

  lstm_policy_kernel<<<Bb / 16, 256, 0, stream>>>(
      state, Wpk, bias,
      mu_w1, mu_b1, mu_w2, mu_b2,
      var_w1, var_b1, var_w2, var_b2,
      val_w1, val_b1, val_w2, val_b2,
      out_mu, out_var, out_val);
}
